// TContextGGANN_39805756899562
// MI455X (gfx1250) — compile-verified, matrix operand order checked by reading, not run on hardware
//
#include <hip/hip_runtime.h>
#include <hip/hip_bf16.h>
#include <math.h>
#include <stdint.h>

// Problem constants (match reference)
#define Dm 256
#define Bm 64
#define Sm 256
#define NLOOP 3
static const long long HN = (long long)Bm * Sm * Dm; // 4,194,304 elements per hidden state

typedef __attribute__((ext_vector_type(16))) __bf16 v16bf;
typedef __attribute__((ext_vector_type(8)))  float  v8f;

// GEMM tile config
#define BM 128
#define BN 64
#define BK 32

// Epilogue modes
#define M_STORE   0  // C = act(acc)
#define M_ACC     1  // C = C + acc
#define M_ACC_ACT 2  // C = act(C + acc)
#define M_ADD_ACT 3  // C = C + act(acc)
// Activations
#define A_NONE  0
#define A_LRELU 1
#define A_SIG   2
#define A_TANH  3
// B operand kinds
#define B_BF16_NK 0  // bf16, pre-transposed [N][K]; staged with async-to-LDS DMA
#define B_F32_NK  1  // fp32, [N][K] (e.g. K-matrix for Q@K^T), cvt while staging
#define B_F32_KN  2  // fp32, [K][N] (e.g. V for A@V), transpose+cvt while staging

__device__ __forceinline__ unsigned short f2bf(float f) {
  unsigned int u = __float_as_uint(f);
  u += 0x7FFFu + ((u >> 16) & 1u);           // round-to-nearest-even
  return (unsigned short)(u >> 16);
}

template<int ACT>
__device__ __forceinline__ float actf(float x) {
  if constexpr (ACT == A_LRELU) return x > 0.f ? x : 0.01f * x;
  else if constexpr (ACT == A_SIG) return 1.f / (1.f + __expf(-x));
  else if constexpr (ACT == A_TANH) return tanhf(x);
  else return x;
}

// ---------------------------------------------------------------------------
// Generic bf16-WMMA GEMM: C[m,n] (+)= act( sum_k (A*Apre)[m,k] * Bt[n,k] )
// Fully specialized on B-kind / epilogue mode / activation / fused premultiply
// so the K-loop between s_barrier and v_wmma is straight-line code.
// ---------------------------------------------------------------------------
template<int BKIND, int MODE, int ACT, bool PRE>
__global__ __launch_bounds__(256)
void wmma_gemm(const float* __restrict__ A, long long lda, long long sA,
               const float* __restrict__ Apre,
               const void* __restrict__ Bp, long long ldb, long long sB,
               float* __restrict__ C, long long ldc, long long sC,
               int K)
{
  __shared__ __align__(16) unsigned short As[BM][BK];
  __shared__ __align__(16) unsigned short Bs[BN][BK];

  const int tid  = threadIdx.x;
  const int lane = tid & 31;            // wave32
  const int wave = tid >> 5;            // 8 waves
  const int wm   = wave >> 1;           // 0..3 (M direction)
  const int wn   = wave & 1;            // 0..1 (N direction)
  const long long batch = blockIdx.z;
  const int m0 = blockIdx.y * BM;
  const int n0 = blockIdx.x * BN;

  const float* Ab  = A + batch * sA;
  const float* Apb = PRE ? (Apre + batch * sA) : nullptr;
  float*       Cb  = C + batch * sC;

  v8f acc[2][2];
#pragma unroll
  for (int i = 0; i < 2; ++i)
#pragma unroll
    for (int j = 0; j < 2; ++j)
#pragma unroll
      for (int e = 0; e < 8; ++e) acc[i][j][e] = 0.f;

  for (int k0 = 0; k0 < K; k0 += BK) {
    if (k0) __syncthreads();

    // ---- stage B tile into Bs[n][k] (contiguous K per row)
    if constexpr (BKIND == B_BF16_NK) {
      // Pure 16B/lane copy of pre-transposed bf16 weights: use the CDNA5
      // async DMA path (ASYNCcnt-tracked, no VGPR round-trip).
      int n  = tid >> 2;
      int ko = (tid & 3) * 8;
      const unsigned short* Bt = (const unsigned short*)Bp + batch * sB;
      unsigned ldsoff = (unsigned)(uintptr_t)&Bs[n][ko];
      unsigned long long ga =
          (unsigned long long)(uintptr_t)&Bt[(long long)(n0 + n) * ldb + k0 + ko];
      asm volatile("global_load_async_to_lds_b128 %0, %1, off"
                   :: "v"(ldsoff), "v"(ga) : "memory");
    } else if constexpr (BKIND == B_F32_NK) {
      int n  = tid >> 2;
      int ko = (tid & 3) * 8;
      const float* Bf = (const float*)Bp + batch * sB;
      float4 x0 = *(const float4*)&Bf[(long long)(n0 + n) * ldb + k0 + ko];
      float4 x1 = *(const float4*)&Bf[(long long)(n0 + n) * ldb + k0 + ko + 4];
      uint4 q;
      q.x = (unsigned)f2bf(x0.x) | ((unsigned)f2bf(x0.y) << 16);
      q.y = (unsigned)f2bf(x0.z) | ((unsigned)f2bf(x0.w) << 16);
      q.z = (unsigned)f2bf(x1.x) | ((unsigned)f2bf(x1.y) << 16);
      q.w = (unsigned)f2bf(x1.z) | ((unsigned)f2bf(x1.w) << 16);
      *(uint4*)&Bs[n][ko] = q;
    } else { // B_F32_KN: transpose while staging
      int n  = tid & 63;
      int kb = (tid >> 6) * 8;
      const float* Bf = (const float*)Bp + batch * sB;
      unsigned short h[8];
#pragma unroll
      for (int k = 0; k < 8; ++k)
        h[k] = f2bf(Bf[(long long)(k0 + kb + k) * ldb + n0 + n]);
      uint4 q;
      q.x = (unsigned)h[0] | ((unsigned)h[1] << 16);
      q.y = (unsigned)h[2] | ((unsigned)h[3] << 16);
      q.z = (unsigned)h[4] | ((unsigned)h[5] << 16);
      q.w = (unsigned)h[6] | ((unsigned)h[7] << 16);
      *(uint4*)&Bs[n][kb] = q;
    }

    // ---- stage A tile: 128x32 fp32 -> bf16 (fused elementwise pre-multiply)
#pragma unroll
    for (int it = 0; it < 4; ++it) {
      int idx = it * 256 + tid;          // 0..1023 float4 slots
      int row = idx >> 3;                // 0..127
      int c4  = (idx & 7) * 4;           // 0..28
      const float* ap = &Ab[(long long)(m0 + row) * lda + k0 + c4];
      float4 v = *(const float4*)ap;
      if constexpr (PRE) {
        const float4 p = *(const float4*)&Apb[(long long)(m0 + row) * lda + k0 + c4];
        v.x *= p.x; v.y *= p.y; v.z *= p.z; v.w *= p.w;
      }
      if (k0 + BK < K) __builtin_prefetch(ap + BK, 0, 1);  // global_prefetch_b8
      uint2 packed;
      packed.x = (unsigned)f2bf(v.x) | ((unsigned)f2bf(v.y) << 16);
      packed.y = (unsigned)f2bf(v.z) | ((unsigned)f2bf(v.w) << 16);
      *(uint2*)&As[row][c4] = packed;
    }

    if constexpr (BKIND == B_BF16_NK) {
      asm volatile("s_wait_asynccnt 0x0" ::: "memory");  // async B-tile landed
    }
    __syncthreads();

    // ---- load fragments (ISA 16-bit A/B layout: two contiguous 8-half runs)
    const int kSel = (lane >> 4) * 8;    // lanes 0-15 -> K base 0, lanes 16-31 -> +8
    union Frag { uint4 q[2]; v16bf v; };
    Frag af[2], bfr[2];
#pragma unroll
    for (int mt = 0; mt < 2; ++mt) {
      int m = wm * 32 + mt * 16 + (lane & 15);
      af[mt].q[0] = *(const uint4*)&As[m][kSel];
      af[mt].q[1] = *(const uint4*)&As[m][16 + kSel];
    }
#pragma unroll
    for (int nt = 0; nt < 2; ++nt) {
      int n = wn * 32 + nt * 16 + (lane & 15);
      bfr[nt].q[0] = *(const uint4*)&Bs[n][kSel];
      bfr[nt].q[1] = *(const uint4*)&Bs[n][16 + kSel];
    }
#pragma unroll
    for (int mt = 0; mt < 2; ++mt)
#pragma unroll
      for (int nt = 0; nt < 2; ++nt)
        acc[mt][nt] = __builtin_amdgcn_wmma_f32_16x16x32_bf16(
            false, af[mt].v, false, bfr[nt].v, (short)0, acc[mt][nt], false, false);
  }

  // ---- epilogue (C/D layout: col = lane&15, row = vgpr + (lane/16)*8)
#pragma unroll
  for (int mt = 0; mt < 2; ++mt) {
#pragma unroll
    for (int nt = 0; nt < 2; ++nt) {
      int col   = n0 + wn * 32 + nt * 16 + (lane & 15);
      int rbase = m0 + wm * 32 + mt * 16 + ((lane >> 4) * 8);
#pragma unroll
      for (int v = 0; v < 8; ++v) {
        long long idx = (long long)(rbase + v) * ldc + col;
        float x = acc[mt][nt][v];
        float o;
        if constexpr (MODE == M_STORE)        o = actf<ACT>(x);
        else if constexpr (MODE == M_ACC)     o = Cb[idx] + x;
        else if constexpr (MODE == M_ACC_ACT) o = actf<ACT>(Cb[idx] + x);
        else                                  o = Cb[idx] + actf<ACT>(x);
        Cb[idx] = o;
      }
    }
  }
}

// ---------------------------------------------------------------------------
// Weight preparation
// ---------------------------------------------------------------------------
__global__ void cvt_bf16(const float* __restrict__ src, unsigned short* __restrict__ dst, int n) {
  int i = blockIdx.x * 256 + threadIdx.x;
  if (i < n) dst[i] = f2bf(src[i]);
}

// dst[c*R + r] = bf16(src[r*srcLd + c])  (transpose + optional column slice)
__global__ void transpose_bf16(const float* __restrict__ src, long long srcLd,
                               unsigned short* __restrict__ dst, int R, int Cc) {
  int i = blockIdx.x * 256 + threadIdx.x;
  if (i >= R * Cc) return;
  int r = i / Cc, c = i % Cc;
  dst[(long long)c * R + r] = f2bf(src[(long long)r * srcLd + c]);
}

// ---------------------------------------------------------------------------
// Elementwise helpers
// ---------------------------------------------------------------------------
__global__ void zero_f4(float4* p, long long n4) {
  long long i = (long long)blockIdx.x * 256 + threadIdx.x;
  if (i < n4) p[i] = make_float4(0.f, 0.f, 0.f, 0.f);
}

__global__ void gru_combine(float* __restrict__ h, const float* __restrict__ z,
                            const float* __restrict__ ht, long long n) {
  long long i = (long long)blockIdx.x * 256 + threadIdx.x;
  if (i < n) {
    float zz = z[i];
    h[i] = (1.f - zz) * h[i] + zz * ht[i];
  }
}

__global__ __launch_bounds__(256)
void softmax_rows(float* __restrict__ s, float scale) {
  __shared__ float red[256];
  float* p = s + (long long)blockIdx.x * 256;
  int t = threadIdx.x;
  float x = p[t] * scale;
  red[t] = x; __syncthreads();
  for (int off = 128; off > 0; off >>= 1) {
    if (t < off) red[t] = fmaxf(red[t], red[t + off]);
    __syncthreads();
  }
  float mx = red[0]; __syncthreads();
  float e = __expf(x - mx);
  red[t] = e; __syncthreads();
  for (int off = 128; off > 0; off >>= 1) {
    if (t < off) red[t] += red[t + off];
    __syncthreads();
  }
  p[t] = e / red[0];
}

// ---------------------------------------------------------------------------
// Host-side launch helper (explicit template instantiations per call site)
// ---------------------------------------------------------------------------
template<int BKIND, int MODE, int ACT, bool PRE>
static inline void launch_gemm(hipStream_t stream,
                               const float* A, long long lda, long long sA,
                               const float* Apre,
                               const void* B, long long ldb, long long sB,
                               float* C, long long ldc, long long sC,
                               int M, int N, int K, int batch) {
  dim3 grid(N / BN, M / BM, batch);
  wmma_gemm<BKIND, MODE, ACT, PRE><<<grid, dim3(256), 0, stream>>>(
      A, lda, sA, Apre, B, ldb, sB, C, ldc, sC, K);
}

// ---------------------------------------------------------------------------
// Host orchestration
// ---------------------------------------------------------------------------
extern "C" void kernel_launch(void* const* d_in, const int* in_sizes, int n_in,
                              void* d_out, int out_size, void* d_ws, size_t ws_size,
                              hipStream_t stream) {
  const float* enc_med    = (const float*)d_in[0];
  const float* enc_inputs = (const float*)d_in[1];
  const float* enc_lab    = (const float*)d_in[2];
  const float* vars_r     = (const float*)d_in[6];   // [4, 2D, D]
  const float* vars_z     = (const float*)d_in[7];   // [4, 2D, 2D]
  const float* vars_out   = (const float*)d_in[8];   // [4, D, D]
  const float* out_h      = (const float*)d_in[9];   // [4, D, D]
  const float* out_x      = (const float*)d_in[10];  // [4, D, D]
  const float* Wq         = (const float*)d_in[11];  // [D, D]
  const float* Wk         = (const float*)d_in[12];  // [D, 3D]
  const float* Wv         = (const float*)d_in[13];  // [D, 3D]

  float* out = (float*)d_out;
  float* he = out;
  float* hl = out + HN;
  float* hi = out + 2 * HN;
  float* hm = out + 3 * HN;

  // --- workspace layout: bf16 weights then fp32 temporaries
  unsigned short* wb = (unsigned short*)d_ws;
  unsigned short* Wout_t = wb;                 // 4 * D*D      (vars_out^T)
  unsigned short* Wr_t   = wb + 262144;        // 4 * D*2D     (vars_r^T)
  unsigned short* Wz_t   = wb + 786432;        // 4 * D*2D     (vars_z[:, :D]^T)
  unsigned short* Woh    = wb + 1310720;       // 4 * D*D      (out_h direct)
  unsigned short* Wox    = wb + 1572864;       // 4 * D*D      (out_x direct)
  unsigned short* Wqb    = wb + 1835008;       // D*D
  unsigned short* Wkb    = wb + 1900544;       // D*3D
  unsigned short* Wvb    = wb + 2097152;       // D*3D
  float* fbase = (float*)((char*)d_ws + 2293760ll * 2);
  float* a_e = fbase;            // reused later as Q
  float* a_l = fbase + HN;       // reused later as K
  float* a_i = fbase + 2 * HN;   // reused later as V
  float* a_m = fbase + 3 * HN;   // reused later as scores [B,S,S]
  float* rb  = fbase + 4 * HN;
  float* zb  = fbase + 5 * HN;
  float* htb = fbase + 6 * HN;

  const int Mfull = Bm * Sm;     // 16384
  dim3 blk(256);

  // --- weight prep
  for (int t = 0; t < 4; ++t) {
    transpose_bf16<<<(Dm * Dm + 255) / 256, blk, 0, stream>>>(
        vars_out + (long long)t * Dm * Dm, Dm, Wout_t + (long long)t * Dm * Dm, Dm, Dm);
    transpose_bf16<<<(2 * Dm * Dm + 255) / 256, blk, 0, stream>>>(
        vars_r + (long long)t * 2 * Dm * Dm, Dm, Wr_t + (long long)t * 2 * Dm * Dm, 2 * Dm, Dm);
    transpose_bf16<<<(2 * Dm * Dm + 255) / 256, blk, 0, stream>>>(
        vars_z + (long long)t * 2 * Dm * 2 * Dm, 2 * Dm, Wz_t + (long long)t * 2 * Dm * Dm, 2 * Dm, Dm);
  }
  cvt_bf16<<<(4 * Dm * Dm + 255) / 256, blk, 0, stream>>>(out_h, Woh, 4 * Dm * Dm);
  cvt_bf16<<<(4 * Dm * Dm + 255) / 256, blk, 0, stream>>>(out_x, Wox, 4 * Dm * Dm);
  cvt_bf16<<<(Dm * Dm + 255) / 256, blk, 0, stream>>>(Wq, Wqb, Dm * Dm);
  cvt_bf16<<<(3 * Dm * Dm + 255) / 256, blk, 0, stream>>>(Wk, Wkb, 3 * Dm * Dm);
  cvt_bf16<<<(3 * Dm * Dm + 255) / 256, blk, 0, stream>>>(Wv, Wvb, 3 * Dm * Dm);

  // --- zero hidden states (live in d_out)
  zero_f4<<<(unsigned)((4 * HN / 4 + 255) / 256), blk, 0, stream>>>((float4*)out, 4 * HN / 4);

  auto gru = [&](float* a, float* h, int ri, int oi) {
    const unsigned short* Wr = Wr_t + (long long)ri * Dm * 2 * Dm;  // [D][2D], ldb=512
    const unsigned short* Wz = Wz_t + (long long)ri * Dm * 2 * Dm;
    const unsigned short* Wh = Woh + (long long)oi * Dm * Dm;       // [D][D]
    const unsigned short* Wx = Wox + (long long)oi * Dm * Dm;
    // r = sigmoid(a @ Wr[:D] + h @ Wr[D:])   (K-split, no concat materialized)
    launch_gemm<B_BF16_NK, M_STORE, A_NONE, false>(stream, a, Dm, 0, nullptr, Wr, 2 * Dm, 0,
                                                   rb, Dm, 0, Mfull, Dm, Dm, 1);
    launch_gemm<B_BF16_NK, M_ACC_ACT, A_SIG, false>(stream, h, Dm, 0, nullptr, Wr + Dm, 2 * Dm, 0,
                                                    rb, Dm, 0, Mfull, Dm, Dm, 1);
    // z = sigmoid(a @ Wz[:D,:D] + h @ Wz[D:,:D])
    launch_gemm<B_BF16_NK, M_STORE, A_NONE, false>(stream, a, Dm, 0, nullptr, Wz, 2 * Dm, 0,
                                                   zb, Dm, 0, Mfull, Dm, Dm, 1);
    launch_gemm<B_BF16_NK, M_ACC_ACT, A_SIG, false>(stream, h, Dm, 0, nullptr, Wz + Dm, 2 * Dm, 0,
                                                    zb, Dm, 0, Mfull, Dm, Dm, 1);
    // h~ = tanh(a @ Wox^T + (r*h) @ Woh^T)   (r*h fused into A staging)
    launch_gemm<B_BF16_NK, M_STORE, A_NONE, false>(stream, a, Dm, 0, nullptr, Wx, Dm, 0,
                                                   htb, Dm, 0, Mfull, Dm, Dm, 1);
    launch_gemm<B_BF16_NK, M_ACC_ACT, A_TANH, true>(stream, rb, Dm, 0, h, Wh, Dm, 0,
                                                    htb, Dm, 0, Mfull, Dm, Dm, 1);
    // h = (1-z)*h + z*h~
    gru_combine<<<(unsigned)((HN + 255) / 256), blk, 0, stream>>>(h, zb, htb, HN);
  };

  for (int loop = 0; loop < NLOOP; ++loop) {
    // messages (enc*h fused into A staging, lrelu in epilogue, sum via ADD_ACT)
    launch_gemm<B_BF16_NK, M_STORE, A_LRELU, true>(stream, enc_lab, Dm, 0, hl,
        Wout_t + 0 * Dm * Dm, Dm, 0, a_e, Dm, 0, Mfull, Dm, Dm, 1);
    launch_gemm<B_BF16_NK, M_ADD_ACT, A_LRELU, true>(stream, enc_inputs, Dm, 0, hi,
        Wout_t + 1 * Dm * Dm, Dm, 0, a_e, Dm, 0, Mfull, Dm, Dm, 1);
    launch_gemm<B_BF16_NK, M_ADD_ACT, A_LRELU, true>(stream, enc_med, Dm, 0, hm,
        Wout_t + 2 * Dm * Dm, Dm, 0, a_e, Dm, 0, Mfull, Dm, Dm, 1);
    launch_gemm<B_BF16_NK, M_STORE, A_LRELU, true>(stream, enc_lab, Dm, 0, he,
        Wout_t + 3 * Dm * Dm, Dm, 0, a_l, Dm, 0, Mfull, Dm, Dm, 1);
    launch_gemm<B_BF16_NK, M_STORE, A_LRELU, true>(stream, enc_inputs, Dm, 0, he,
        Wout_t + 3 * Dm * Dm, Dm, 0, a_i, Dm, 0, Mfull, Dm, Dm, 1);
    launch_gemm<B_BF16_NK, M_STORE, A_LRELU, true>(stream, enc_med, Dm, 0, he,
        Wout_t + 3 * Dm * Dm, Dm, 0, a_m, Dm, 0, Mfull, Dm, Dm, 1);
    // GGNN gated updates
    gru(a_e, he, 3, 0);
    gru(a_l, hl, 0, 1);
    gru(a_i, hi, 1, 2);
    gru(a_m, hm, 2, 3);
  }

  // --- cross-type attention
  float* Qb = a_e; float* Kb = a_l; float* Vb = a_i; float* sc = a_m;
  launch_gemm<B_BF16_NK, M_STORE, A_NONE, false>(stream, he, Dm, 0, nullptr, Wqb, Dm, 0,
                                                 Qb, Dm, 0, Mfull, Dm, Dm, 1);
  // K = h_l@Wk[:,:D]^T + h_i@Wk[:,D:2D]^T + h_m@Wk[:,2D:]^T  (K-split over concat)
  launch_gemm<B_BF16_NK, M_STORE, A_NONE, false>(stream, hl, Dm, 0, nullptr, Wkb + 0 * Dm, 3 * Dm, 0,
                                                 Kb, Dm, 0, Mfull, Dm, Dm, 1);
  launch_gemm<B_BF16_NK, M_ACC, A_NONE, false>(stream, hi, Dm, 0, nullptr, Wkb + 1 * Dm, 3 * Dm, 0,
                                               Kb, Dm, 0, Mfull, Dm, Dm, 1);
  launch_gemm<B_BF16_NK, M_ACC, A_NONE, false>(stream, hm, Dm, 0, nullptr, Wkb + 2 * Dm, 3 * Dm, 0,
                                               Kb, Dm, 0, Mfull, Dm, Dm, 1);
  launch_gemm<B_BF16_NK, M_STORE, A_NONE, false>(stream, hl, Dm, 0, nullptr, Wvb + 0 * Dm, 3 * Dm, 0,
                                                 Vb, Dm, 0, Mfull, Dm, Dm, 1);
  launch_gemm<B_BF16_NK, M_ACC, A_NONE, false>(stream, hi, Dm, 0, nullptr, Wvb + 1 * Dm, 3 * Dm, 0,
                                               Vb, Dm, 0, Mfull, Dm, Dm, 1);
  launch_gemm<B_BF16_NK, M_ACC, A_NONE, false>(stream, hm, Dm, 0, nullptr, Wvb + 2 * Dm, 3 * Dm, 0,
                                               Vb, Dm, 0, Mfull, Dm, Dm, 1);
  // scores = Q @ K^T per batch (K matrix is already [N=S][K=D] layout for us)
  launch_gemm<B_F32_NK, M_STORE, A_NONE, false>(stream, Qb, Dm, (long long)Sm * Dm, nullptr,
      Kb, Dm, (long long)Sm * Dm, sc, Sm, (long long)Sm * Sm, Sm, Sm, Dm, Bm);
  softmax_rows<<<Bm * Sm, blk, 0, stream>>>(sc, 1.0f / 16.0f);  // 1/sqrt(D)
  // h_e += A @ V per batch (V is [K=S][N=D]; transpose-staged)
  launch_gemm<B_F32_KN, M_ACC, A_NONE, false>(stream, sc, Sm, (long long)Sm * Sm, nullptr,
      Vb, Dm, (long long)Sm * Dm, he, Dm, (long long)Sm * Dm, Sm, Dm, Sm, Bm);
}
